// MultiHeadSelfAttention_8211977470458
// MI455X (gfx1250) — compile-verified, matrix operand order checked by reading, not run on hardware
//
#include <hip/hip_runtime.h>
#include <hip/hip_bf16.h>

// ---------------------------------------------------------------------------
// MultiHeadSelfAttention (with the reference's mislabeled einsum semantics)
// for MI455X / gfx1250. Compute-bound in 4x (8192x1024x1024) GEMMs -> use
// v_wmma_f32_16x16x32_bf16 with 3-term bf16 hi/lo split for ~f32 accuracy.
// GEMM tiles staged via GLOBAL_LOAD_ASYNC_TO_LDS_B128 (ASYNCcnt) with
// double-buffered LDS so tile prefetch overlaps the WMMA inner loop.
// ---------------------------------------------------------------------------

typedef __attribute__((ext_vector_type(16))) __bf16 v16bf;
typedef __attribute__((ext_vector_type(8)))  __bf16 v8bf;
typedef __attribute__((ext_vector_type(8)))  float  v8f;

#define NB   4
#define SEQ  2048
#define EMB  1024
#define HDS  16
#define HDIM 64
#define TOK  (NB * SEQ)      // 8192

// ------------------------------ split f32 -> bf16 hi/lo ---------------------
__global__ void split_kernel(const float* __restrict__ src,
                             __bf16* __restrict__ hi, __bf16* __restrict__ lo,
                             int n) {
    int i = blockIdx.x * blockDim.x + threadIdx.x;
    if (i >= n) return;
    float x = src[i];
    __bf16 h = (__bf16)x;
    hi[i] = h;
    lo[i] = (__bf16)(x - (float)h);
}

// ------------------------------ WMMA GEMM -----------------------------------
// C[M,N] = A[M,K] * B[N,K]^T (+bias).  A,B given as bf16 (hi,lo) pairs.
// Block tile 128x128, 256 threads = 8 waves (wave32), wave tile 32x64.
// Fragment packing assumption (CDNA5 ISA 7.12.2, 16-bit A 16x32):
//   lanes 0-15 : row M=lane,    K elements [c..c+7] then [c+16..c+23]
//   lanes 16-31: row M=lane-16, K elements [c+8..c+15] then [c+24..c+31]
// B is supplied N-major symmetric to A (i.e. W rows directly).
__global__ __launch_bounds__(256)
void gemm_bf16x3(const __bf16* __restrict__ Ahi, const __bf16* __restrict__ Alo,
                 const __bf16* __restrict__ Bhi, const __bf16* __restrict__ Blo,
                 float* __restrict__ C, const float* __restrict__ bias,
                 int M, int N, int K) {
    // Double-buffered LDS tiles: 128 rows x 32 cols, row stride 40 (+16B pad)
    __shared__ __attribute__((aligned(16))) __bf16 sAhi[2][128 * 40];
    __shared__ __attribute__((aligned(16))) __bf16 sAlo[2][128 * 40];
    __shared__ __attribute__((aligned(16))) __bf16 sBhi[2][128 * 40];
    __shared__ __attribute__((aligned(16))) __bf16 sBlo[2][128 * 40];

    const int tid   = threadIdx.x;
    const int lane  = tid & 31;
    const int wid   = tid >> 5;
    const int waveM = wid & 3;      // 4 waves along M
    const int waveN = wid >> 2;     // 2 waves along N
    const int blockM = blockIdx.y * 128;
    const int blockN = blockIdx.x * 128;

    v8f acc[2][4];
#pragma unroll
    for (int mt = 0; mt < 2; ++mt)
#pragma unroll
        for (int nt = 0; nt < 4; ++nt)
#pragma unroll
            for (int r = 0; r < 8; ++r) acc[mt][nt][r] = 0.0f;

    const int lrow = tid >> 1;          // 0..127
    const int lseg = tid & 1;           // 16-col segment
    const int half = lane >> 4;         // 0|1 : which K-half this lane packs
    const int frow = lane & 15;
    const int ld   = lrow * 40 + lseg * 16;

    // Async global->LDS 16B move. LDS dest = low 32 bits of the generic
    // pointer to shared memory (== AS(3) byte offset on amdgcn).
    auto issue16 = [](const __bf16* g, const __bf16* l) {
        unsigned loff = (unsigned)(unsigned long long)(const void*)l;
        asm volatile("global_load_async_to_lds_b128 %0, %1, off"
                     :: "v"(loff), "v"(g) : "memory");
    };
    auto stage_issue = [&](int k0, int st) {
        const long gA = (long)(blockM + lrow) * K + k0 + lseg * 16;
        const long gB = (long)(blockN + lrow) * K + k0 + lseg * 16;
        issue16(Ahi + gA,     &sAhi[st][ld]);
        issue16(Ahi + gA + 8, &sAhi[st][ld + 8]);
        issue16(Alo + gA,     &sAlo[st][ld]);
        issue16(Alo + gA + 8, &sAlo[st][ld + 8]);
        issue16(Bhi + gB,     &sBhi[st][ld]);
        issue16(Bhi + gB + 8, &sBhi[st][ld + 8]);
        issue16(Blo + gB,     &sBlo[st][ld]);
        issue16(Blo + gB + 8, &sBlo[st][ld + 8]);
    };

    const int steps = K >> 5;           // K/32 k-steps
    stage_issue(0, 0);                  // prologue: stages 0 and 1 in flight
    if (steps > 1) stage_issue(32, 1);

    for (int i = 0; i < steps; ++i) {
        const int st = i & 1;
        // 8 async ops per stage per wave, completing in order: waiting to 8
        // guarantees the current stage is resident while the next streams in.
        if (i + 1 < steps)
            asm volatile("s_wait_asynccnt 0x8" ::: "memory");
        else
            asm volatile("s_wait_asynccnt 0x0" ::: "memory");
        __syncthreads();                // stage st visible to all waves

        v16bf afh[2], afl[2], bfh[4], bfl[4];
#pragma unroll
        for (int mt = 0; mt < 2; ++mt) {
            const int row = waveM * 32 + mt * 16 + frow;
            const int o0  = row * 40 + half * 8;
            *((v8bf*)&afh[mt])     = *(const v8bf*)&sAhi[st][o0];
            *((v8bf*)&afh[mt] + 1) = *(const v8bf*)&sAhi[st][o0 + 16];
            *((v8bf*)&afl[mt])     = *(const v8bf*)&sAlo[st][o0];
            *((v8bf*)&afl[mt] + 1) = *(const v8bf*)&sAlo[st][o0 + 16];
        }
#pragma unroll
        for (int nt = 0; nt < 4; ++nt) {
            const int row = waveN * 64 + nt * 16 + frow;
            const int o0  = row * 40 + half * 8;
            *((v8bf*)&bfh[nt])     = *(const v8bf*)&sBhi[st][o0];
            *((v8bf*)&bfh[nt] + 1) = *(const v8bf*)&sBhi[st][o0 + 16];
            *((v8bf*)&bfl[nt])     = *(const v8bf*)&sBlo[st][o0];
            *((v8bf*)&bfl[nt] + 1) = *(const v8bf*)&sBlo[st][o0 + 16];
        }
#pragma unroll
        for (int mt = 0; mt < 2; ++mt)
#pragma unroll
            for (int nt = 0; nt < 4; ++nt) {
                acc[mt][nt] = __builtin_amdgcn_wmma_f32_16x16x32_bf16(
                    false, afh[mt], false, bfh[nt], (short)0, acc[mt][nt], false, false);
                acc[mt][nt] = __builtin_amdgcn_wmma_f32_16x16x32_bf16(
                    false, afh[mt], false, bfl[nt], (short)0, acc[mt][nt], false, false);
                acc[mt][nt] = __builtin_amdgcn_wmma_f32_16x16x32_bf16(
                    false, afl[mt], false, bfh[nt], (short)0, acc[mt][nt], false, false);
            }

        __syncthreads();                // all waves done reading stage st
        if (i + 2 < steps)
            stage_issue((i + 2) << 5, st);  // safe to overwrite stage st
    }

    // Epilogue: C/D layout — VGPR r: lanes 0-15 -> M=r, lanes 16-31 -> M=r+8
#pragma unroll
    for (int mt = 0; mt < 2; ++mt)
#pragma unroll
        for (int nt = 0; nt < 4; ++nt) {
            const int col   = blockN + waveN * 64 + nt * 16 + frow;
            const int rbase = blockM + waveM * 32 + mt * 16 + half * 8;
            const float b   = bias ? bias[col] : 0.0f;
#pragma unroll
            for (int r = 0; r < 8; ++r)
                C[(long)(rbase + r) * N + col] = acc[mt][nt][r] + b;
        }
}

// ------------------------------ head-Gram energy ----------------------------
// G[n,qh,kh,s] = 0.125 * sum_d Q[t, qh*64+d] * K[t, kh*64+d],  t = n*2048+s
__global__ __launch_bounds__(64)
void energy_kernel(const float* __restrict__ Q, const float* __restrict__ K,
                   float* __restrict__ G) {
    __shared__ float qs[EMB], ks[EMB];
    const int t = blockIdx.x, n = t >> 11, s = t & 2047;
    for (int i = threadIdx.x; i < EMB; i += 64) {
        qs[i] = Q[(long)t * EMB + i];
        ks[i] = K[(long)t * EMB + i];
    }
    __syncthreads();
    const int p0 = threadIdx.x * 4;
    for (int p = p0; p < p0 + 4; ++p) {
        const int qh = p >> 4, kh = p & 15;
        const float* qp = qs + qh * HDIM;
        const float* kp = ks + kh * HDIM;
        float a = 0.0f;
#pragma unroll
        for (int d = 0; d < HDIM; ++d) a += qp[d] * kp[d];
        G[(long)(((n * HDS + qh) * HDS + kh) << 11) + s] = a * 0.125f;
    }
}

// ---------------- softmax stats over the seq axis (per G row) ---------------
__global__ __launch_bounds__(256)
void stats_kernel(const float* __restrict__ G, float* __restrict__ rmax,
                  float* __restrict__ rz) {
    __shared__ float red[256];
    const int row = blockIdx.x;
    const float* g = G + (long)row * SEQ;
    float m = -1e30f;
    for (int i = threadIdx.x; i < SEQ; i += 256) m = fmaxf(m, g[i]);
    red[threadIdx.x] = m; __syncthreads();
    for (int s = 128; s > 0; s >>= 1) {
        if (threadIdx.x < s) red[threadIdx.x] = fmaxf(red[threadIdx.x], red[threadIdx.x + s]);
        __syncthreads();
    }
    m = red[0]; __syncthreads();
    float z = 0.0f;
    for (int i = threadIdx.x; i < SEQ; i += 256) z += __expf(g[i] - m);
    red[threadIdx.x] = z; __syncthreads();
    for (int s = 128; s > 0; s >>= 1) {
        if (threadIdx.x < s) red[threadIdx.x] += red[threadIdx.x + s];
        __syncthreads();
    }
    if (threadIdx.x == 0) { rmax[row] = m; rz[row] = red[0]; }
}

// Aw[n,qh,s] = sum_kh softmax(G)[n,qh,kh,s]
__global__ void aw_kernel(const float* __restrict__ G, const float* __restrict__ rmax,
                          const float* __restrict__ rz, float* __restrict__ Aw) {
    const int idx = blockIdx.x * blockDim.x + threadIdx.x;   // < 4*16*2048
    if (idx >= NB * HDS * SEQ) return;
    const int s = idx & 2047, qh = (idx >> 11) & 15, n = idx >> 15;
    const int rbase = (n * HDS + qh) * HDS;
    const float* gp = G + (long)rbase * SEQ + s;
    float a = 0.0f;
#pragma unroll
    for (int kh = 0; kh < HDS; ++kh)
        a += __expf(gp[(long)kh * SEQ] - rmax[rbase + kh]) / rz[rbase + kh];
    Aw[idx] = a;
}

// Vs[t,d] = sum_h V[t, h*64+d]
__global__ void vs_kernel(const float* __restrict__ V, float* __restrict__ Vs) {
    const int idx = blockIdx.x * blockDim.x + threadIdx.x;   // < 8192*64
    if (idx >= TOK * HDIM) return;
    const int d = idx & 63, t = idx >> 6;
    float a = 0.0f;
#pragma unroll
    for (int h = 0; h < HDS; ++h) a += V[(long)t * EMB + h * HDIM + d];
    Vs[idx] = a;
}

// U[t',e'] = Aw[n,qh,s] * Vs[n,s,d], raw-flatten: t'=n*2048+qh*128+s/16,
// e'=(s%16)*64+d.  Emitted directly as bf16 hi/lo for the final GEMM.
__global__ void build_u_kernel(const float* __restrict__ Aw, const float* __restrict__ Vs,
                               __bf16* __restrict__ Uhi, __bf16* __restrict__ Ulo) {
    const int idx = blockIdx.x * blockDim.x + threadIdx.x;   // < 2^23
    const int d = idx & 63, s = (idx >> 6) & 2047, qh = (idx >> 17) & 15, n = idx >> 21;
    const float val = Aw[((n * HDS + qh) << 11) + s] * Vs[(((n << 11) + s) << 6) + d];
    const long dst = (long)((n << 11) + (qh << 7) + (s >> 4)) * EMB + ((s & 15) << 6) + d;
    const __bf16 h = (__bf16)val;
    Uhi[dst] = h;
    Ulo[dst] = (__bf16)(val - (float)h);
}

// ------------------------------ host orchestration --------------------------
extern "C" void kernel_launch(void* const* d_in, const int* in_sizes, int n_in,
                              void* d_out, int out_size, void* d_ws, size_t ws_size,
                              hipStream_t stream) {
    const float* x  = (const float*)d_in[0];
    const float* Wq = (const float*)d_in[1];
    const float* Wk = (const float*)d_in[2];
    const float* Wv = (const float*)d_in[3];
    const float* Wo = (const float*)d_in[4];
    const float* bo = (const float*)d_in[5];

    char* p = (char*)d_ws;
    auto carve = [&](size_t bytes) { void* r = (void*)p; p += (bytes + 255) & ~(size_t)255; return r; };

    const size_t XE = (size_t)TOK * EMB;      // 8388608
    const size_t WE = (size_t)EMB * EMB;      // 1048576

    __bf16* Xhi = (__bf16*)carve(XE * 2);
    __bf16* Xlo = (__bf16*)carve(XE * 2);
    __bf16* Whi[4], *Wlo[4];
    for (int i = 0; i < 4; ++i) {
        Whi[i] = (__bf16*)carve(WE * 2);
        Wlo[i] = (__bf16*)carve(WE * 2);
    }
    float* Q  = (float*)carve(XE * 4);        // later aliased by Uhi/Ulo
    float* Kb = (float*)carve(XE * 4);
    float* V  = (float*)carve(XE * 4);
    float* G  = (float*)carve((size_t)NB * HDS * HDS * SEQ * 4);
    float* rmax = (float*)carve(NB * HDS * HDS * 4);
    float* rz   = (float*)carve(NB * HDS * HDS * 4);
    float* Aw = (float*)carve((size_t)NB * HDS * SEQ * 4);
    float* Vs = (float*)carve((size_t)TOK * HDIM * 4);
    __bf16* Uhi = (__bf16*)Q;                 // 16MB each inside Q's 32MB
    __bf16* Ulo = (__bf16*)Q + XE;

    // 1) split inputs to bf16 hi/lo
    split_kernel<<<(int)(XE / 256), 256, 0, stream>>>(x, Xhi, Xlo, (int)XE);
    const float* Ws[4] = {Wq, Wk, Wv, Wo};
    for (int i = 0; i < 4; ++i)
        split_kernel<<<(int)(WE / 256), 256, 0, stream>>>(Ws[i], Whi[i], Wlo[i], (int)WE);

    // 2) Q,K,V projections: 8192x1024 @ (1024x1024)^T via 3x-bf16 WMMA
    dim3 ggrid(EMB / 128, TOK / 128);
    gemm_bf16x3<<<ggrid, 256, 0, stream>>>(Xhi, Xlo, Whi[0], Wlo[0], Q,  nullptr, TOK, EMB, EMB);
    gemm_bf16x3<<<ggrid, 256, 0, stream>>>(Xhi, Xlo, Whi[1], Wlo[1], Kb, nullptr, TOK, EMB, EMB);
    gemm_bf16x3<<<ggrid, 256, 0, stream>>>(Xhi, Xlo, Whi[2], Wlo[2], V,  nullptr, TOK, EMB, EMB);

    // 3) per-token 16x16 head-Gram (scaled by 1/sqrt(64))
    energy_kernel<<<TOK, 64, 0, stream>>>(Q, Kb, G);

    // 4) softmax stats over seq axis (1024 rows of 2048)
    stats_kernel<<<NB * HDS * HDS, 256, 0, stream>>>(G, rmax, rz);

    // 5) Aw = sum_kh softmax ; Vs = sum_h v
    aw_kernel<<<(NB * HDS * SEQ) / 256, 256, 0, stream>>>(G, rmax, rz, Aw);
    vs_kernel<<<(TOK * HDIM) / 256, 256, 0, stream>>>(V, Vs);

    // 6) build reshaped U (overwrites Q buffer; Q no longer needed)
    build_u_kernel<<<(int)(XE / 256), 256, 0, stream>>>(Aw, Vs, Uhi, Ulo);

    // 7) output projection with bias -> d_out
    gemm_bf16x3<<<ggrid, 256, 0, stream>>>(Uhi, Ulo, Whi[3], Wlo[3], (float*)d_out, bo, TOK, EMB, EMB);
}
